// GNNModel_51977694216572
// MI455X (gfx1250) — compile-verified, matrix operand order checked by reading, not run on hardware
//
#include <hip/hip_runtime.h>

#define N_NODES 50000
#define N_EDGES 600000
#define IN_DIM 128
#define HID 256
#define N_CLASSES 16

typedef __attribute__((ext_vector_type(16))) __bf16 v16bf;
typedef __attribute__((ext_vector_type(8)))  float  v8f;
typedef __attribute__((ext_vector_type(8)))  unsigned int v8u;

// float -> bf16 bits, round-to-nearest-even
__device__ __forceinline__ unsigned short f2bf(float f) {
    unsigned u = __builtin_bit_cast(unsigned, f);
    unsigned r = u + 0x7FFFu + ((u >> 16) & 1u);
    return (unsigned short)(r >> 16);
}

// ---------------------------------------------------------------------------
// Edge scatter: one wave32 per edge. F=128: lane covers 4 floats (float4).
// Accumulates neighbor sums + degree via f32 atomics (resolved in L2).
// ---------------------------------------------------------------------------
__global__ __launch_bounds__(256)
void scatter_sum_128(const float* __restrict__ x, const int* __restrict__ ei,
                     float* __restrict__ agg, float* __restrict__ deg) {
    int gid  = blockIdx.x * blockDim.x + threadIdx.x;
    int e    = gid >> 5;
    int lane = gid & 31;
    if (e >= N_EDGES) return;
    int src = ei[e];
    int dst = ei[N_EDGES + e];
    const float4 v = *(const float4*)(x + (size_t)src * IN_DIM + lane * 4);
    float* a = agg + (size_t)dst * IN_DIM + lane * 4;
    atomicAdd(a + 0, v.x); atomicAdd(a + 1, v.y);
    atomicAdd(a + 2, v.z); atomicAdd(a + 3, v.w);
    if (lane == 0) atomicAdd(deg + dst, 1.0f);
}

// F=256: lane covers 8 floats (2x float4). Degree already computed.
__global__ __launch_bounds__(256)
void scatter_sum_256(const float* __restrict__ h, const int* __restrict__ ei,
                     float* __restrict__ agg) {
    int gid  = blockIdx.x * blockDim.x + threadIdx.x;
    int e    = gid >> 5;
    int lane = gid & 31;
    if (e >= N_EDGES) return;
    int src = ei[e];
    int dst = ei[N_EDGES + e];
    const float4* p = (const float4*)(h + (size_t)src * HID + lane * 8);
    float4 v0 = p[0], v1 = p[1];
    float* a = agg + (size_t)dst * HID + lane * 8;
    atomicAdd(a + 0, v0.x); atomicAdd(a + 1, v0.y);
    atomicAdd(a + 2, v0.z); atomicAdd(a + 3, v0.w);
    atomicAdd(a + 4, v1.x); atomicAdd(a + 5, v1.y);
    atomicAdd(a + 6, v1.z); atomicAdd(a + 7, v1.w);
}

__global__ __launch_bounds__(256)
void make_invdeg(float* __restrict__ deg, int n) {
    int i = blockIdx.x * blockDim.x + threadIdx.x;
    if (i < n) deg[i] = 1.0f / fmaxf(deg[i], 1.0f);
}

// ---------------------------------------------------------------------------
// Fused SAGE concat-GEMM:  out = act( [agg*invdeg ‖ self] @ [wl;wr] + bias )
// Block: 256 threads = 8 waves, tile 64(M) x 64(N), K stepped by 32.
// Wave w -> (mw=w&3, nw=w>>2): 16x32 output = two v_wmma_f32_16x16x32_bf16.
// A/B staged in LDS as bf16 (converted during cooperative load).
// ---------------------------------------------------------------------------
template <int RELU>
__global__ __launch_bounds__(256)
void sage_gemm(const float* __restrict__ agg,   // [M,Ka] neighbor sums
               const float* __restrict__ selfx, // [M,Ks]
               const float* __restrict__ invd,  // [M]
               const float* __restrict__ wl,    // [Ka,N]
               const float* __restrict__ wr,    // [Ks,N]
               const float* __restrict__ bias,  // [N]
               float* __restrict__ out,         // [M,N]
               int M, int Ka, int Ks, int N, int tilesN) {
    const int bx = blockIdx.x;
    const int n_base = (bx % tilesN) * 64;
    const int m_base = (bx / tilesN) * 64;
    const int tid  = threadIdx.x;
    const int w    = tid >> 5, lane = tid & 31;
    const int mw   = w & 3,  nw   = w >> 2;
    const int half = lane >> 4, mr = lane & 15;

    __shared__ unsigned short As[64][40];  // [m][k], stride 40 keeps b32 alignment
    __shared__ unsigned short Bs[64][40];  // [n][k]

    v8f acc0 = {}; v8f acc1 = {};
    const int K = Ka + Ks;

    for (int k0 = 0; k0 < K; k0 += 32) {
        // ---- stage A tile (64x32): thread -> row=tid>>2, 8 consecutive k
        {
            int row = tid >> 2, kseg = (tid & 3) * 8;
            int m = m_base + row;
            float vals[8] = {0,0,0,0,0,0,0,0};
            if (m < M) {
                int kg = k0 + kseg;
                if (kg < Ka) {
                    float iv = invd[m];
                    const float4* p = (const float4*)(agg + (size_t)m * Ka + kg);
                    float4 a0 = p[0], a1 = p[1];
                    vals[0]=a0.x*iv; vals[1]=a0.y*iv; vals[2]=a0.z*iv; vals[3]=a0.w*iv;
                    vals[4]=a1.x*iv; vals[5]=a1.y*iv; vals[6]=a1.z*iv; vals[7]=a1.w*iv;
                } else {
                    const float4* p = (const float4*)(selfx + (size_t)m * Ks + (kg - Ka));
                    float4 a0 = p[0], a1 = p[1];
                    vals[0]=a0.x; vals[1]=a0.y; vals[2]=a0.z; vals[3]=a0.w;
                    vals[4]=a1.x; vals[5]=a1.y; vals[6]=a1.z; vals[7]=a1.w;
                }
            }
            #pragma unroll
            for (int j = 0; j < 8; ++j) As[row][kseg + j] = f2bf(vals[j]);
        }
        // ---- stage B tile (32x64) transposed to [n][k]: thread -> k=tid>>3
        {
            int kk = tid >> 3, nseg = (tid & 7) * 8;
            int kg = k0 + kk;
            const float* wp = (kg < Ka) ? (wl + (size_t)kg * N + n_base + nseg)
                                        : (wr + (size_t)(kg - Ka) * N + n_base + nseg);
            const float4* p = (const float4*)wp;
            float4 b0 = p[0], b1 = p[1];
            float vals[8] = {b0.x,b0.y,b0.z,b0.w,b1.x,b1.y,b1.z,b1.w};
            #pragma unroll
            for (int j = 0; j < 8; ++j) Bs[nseg + j][kk] = f2bf(vals[j]);
        }
        __syncthreads();

        // ---- gather fragments per ISA 16-bit A/B layout, issue WMMA
        v8u au, bu0, bu1;
        #pragma unroll
        for (int i = 0; i < 8; ++i) {
            int kk = ((i >> 2) << 4) + (half << 3) + ((i & 3) << 1);
            au[i]  = *(const unsigned int*)&As[mw * 16 + mr][kk];
            bu0[i] = *(const unsigned int*)&Bs[nw * 32 + mr][kk];
            bu1[i] = *(const unsigned int*)&Bs[nw * 32 + 16 + mr][kk];
        }
        v16bf a  = __builtin_bit_cast(v16bf, au);
        v16bf b0 = __builtin_bit_cast(v16bf, bu0);
        v16bf b1 = __builtin_bit_cast(v16bf, bu1);
        acc0 = __builtin_amdgcn_wmma_f32_16x16x32_bf16(false, a, false, b0,
                                                       (short)0, acc0, false, false);
        acc1 = __builtin_amdgcn_wmma_f32_16x16x32_bf16(false, a, false, b1,
                                                       (short)0, acc1, false, false);
        __syncthreads();
    }

    // ---- epilogue: bias + optional ReLU.  C layout: vgpr r -> m = half*8+r, n = mr
    int n0 = n_base + nw * 32 + mr;
    int n1 = n0 + 16;
    float bb0 = bias[n0], bb1 = bias[n1];
    #pragma unroll
    for (int r = 0; r < 8; ++r) {
        int m = m_base + mw * 16 + half * 8 + r;
        if (m < M) {
            float v0 = acc0[r] + bb0;
            float v1 = acc1[r] + bb1;
            if (RELU) { v0 = fmaxf(v0, 0.0f); v1 = fmaxf(v1, 0.0f); }
            out[(size_t)m * N + n0] = v0;
            out[(size_t)m * N + n1] = v1;
        }
    }
}

// ---------------------------------------------------------------------------
// Classifier: [M,256] @ [256,16] + bc.  N=16 = exactly one WMMA column tile.
// Block covers 128 rows (8 waves x 16). wc staged once in LDS.
// ---------------------------------------------------------------------------
__global__ __launch_bounds__(256)
void classifier_gemm(const float* __restrict__ h,  // [M,HID]
                     const float* __restrict__ wc, // [HID,16]
                     const float* __restrict__ bc, // [16]
                     float* __restrict__ out, int M) {
    const int tid  = threadIdx.x;
    const int w    = tid >> 5, lane = tid & 31;
    const int half = lane >> 4, mr = lane & 15;
    const int m_base = blockIdx.x * 128;

    __shared__ unsigned short As[128][40];
    __shared__ unsigned short Bs[16][264];   // whole wc, [n][k], b32-aligned stride

    {   // stage B once: thread tid -> k row of wc, all 16 n
        const float4* p = (const float4*)(wc + (size_t)tid * N_CLASSES);
        float4 c0 = p[0], c1 = p[1], c2 = p[2], c3 = p[3];
        float vals[16] = {c0.x,c0.y,c0.z,c0.w, c1.x,c1.y,c1.z,c1.w,
                          c2.x,c2.y,c2.z,c2.w, c3.x,c3.y,c3.z,c3.w};
        #pragma unroll
        for (int n = 0; n < 16; ++n) Bs[n][tid] = f2bf(vals[n]);
    }

    v8f acc = {};
    for (int k0 = 0; k0 < HID; k0 += 32) {
        __syncthreads();
        {   // stage A slice (128x32): thread -> row=tid>>1, 16 consecutive k
            int row = tid >> 1, kseg = (tid & 1) * 16;
            int m = m_base + row;
            float vals[16] = {0};
            if (m < M) {
                const float4* p = (const float4*)(h + (size_t)m * HID + k0 + kseg);
                float4 a0 = p[0], a1 = p[1], a2 = p[2], a3 = p[3];
                float t[16] = {a0.x,a0.y,a0.z,a0.w, a1.x,a1.y,a1.z,a1.w,
                               a2.x,a2.y,a2.z,a2.w, a3.x,a3.y,a3.z,a3.w};
                #pragma unroll
                for (int j = 0; j < 16; ++j) vals[j] = t[j];
            }
            #pragma unroll
            for (int j = 0; j < 16; ++j) As[row][kseg + j] = f2bf(vals[j]);
        }
        __syncthreads();

        v8u au, bu;
        #pragma unroll
        for (int i = 0; i < 8; ++i) {
            int kk = ((i >> 2) << 4) + (half << 3) + ((i & 3) << 1);
            au[i] = *(const unsigned int*)&As[w * 16 + mr][kk];
            bu[i] = *(const unsigned int*)&Bs[mr][k0 + kk];
        }
        v16bf a = __builtin_bit_cast(v16bf, au);
        v16bf b = __builtin_bit_cast(v16bf, bu);
        acc = __builtin_amdgcn_wmma_f32_16x16x32_bf16(false, a, false, b,
                                                      (short)0, acc, false, false);
    }

    float bb = bc[mr];
    #pragma unroll
    for (int r = 0; r < 8; ++r) {
        int m = m_base + w * 16 + half * 8 + r;
        if (m < M) out[(size_t)m * N_CLASSES + mr] = acc[r] + bb;
    }
}

// ---------------------------------------------------------------------------
extern "C" void kernel_launch(void* const* d_in, const int* in_sizes, int n_in,
                              void* d_out, int out_size, void* d_ws, size_t ws_size,
                              hipStream_t stream) {
    const float* x    = (const float*)d_in[0];
    const int*   ei   = (const int*)  d_in[1];   // edge_index [2, E]
    const float* w1_l = (const float*)d_in[2];
    const float* b1_l = (const float*)d_in[3];
    const float* w1_r = (const float*)d_in[4];
    const float* w2_l = (const float*)d_in[5];
    const float* b2_l = (const float*)d_in[6];
    const float* w2_r = (const float*)d_in[7];
    const float* wc   = (const float*)d_in[8];
    const float* bc   = (const float*)d_in[9];
    float* out = (float*)d_out;

    // workspace layout (floats): deg | agg1 | agg2 | h1 | h2   (~180 MB)
    float* ws = (float*)d_ws;
    const size_t OFF_DEG  = 0;
    const size_t OFF_AGG1 = 50176;                               // deg padded
    const size_t OFF_AGG2 = OFF_AGG1 + (size_t)N_NODES * IN_DIM; // 6.4M
    const size_t OFF_H1   = OFF_AGG2 + (size_t)N_NODES * HID;    // +12.8M
    const size_t OFF_H2   = OFF_H1   + (size_t)N_NODES * HID;
    float* deg  = ws + OFF_DEG;
    float* agg1 = ws + OFF_AGG1;
    float* agg2 = ws + OFF_AGG2;
    float* h1   = ws + OFF_H1;
    float* h2   = ws + OFF_H2;

    // zero deg + agg1 + agg2 (accumulators) every call
    hipMemsetAsync(ws, 0, OFF_H1 * sizeof(float), stream);

    const int edgeBlocks = (N_EDGES * 32) / 256;           // 75000
    const int tilesM     = (N_NODES + 63) / 64;            // 782
    const dim3 gemmGrid(tilesM * 4);                        // N=256 -> 4 n-tiles

    // layer 1
    scatter_sum_128<<<edgeBlocks, 256, 0, stream>>>(x, ei, agg1, deg);
    make_invdeg<<<(N_NODES + 255) / 256, 256, 0, stream>>>(deg, N_NODES);
    sage_gemm<1><<<gemmGrid, 256, 0, stream>>>(agg1, x, deg, w1_l, w1_r, b1_l,
                                               h1, N_NODES, IN_DIM, IN_DIM, HID, 4);
    // layer 2
    scatter_sum_256<<<edgeBlocks, 256, 0, stream>>>(h1, ei, agg2);
    sage_gemm<1><<<gemmGrid, 256, 0, stream>>>(agg2, h1, deg, w2_l, w2_r, b2_l,
                                               h2, N_NODES, HID, HID, HID, 4);
    // classifier
    classifier_gemm<<<(N_NODES + 127) / 128, 256, 0, stream>>>(h2, wc, bc, out, N_NODES);
}
